// ControllerNetwork_22711787061771
// MI455X (gfx1250) — matrix operand order repaired and checked by reading, not compiled, not run on hardware
//
#include <hip/hip_runtime.h>
#include <cstdint>

// ---------------------------------------------------------------------------
// ControllerNetwork (2-layer LSTM controller, H=E=1024, T=64) for gfx1250.
// Persistent single-kernel design: 32 WGs x 256 threads, grid-wide sense
// barrier, bf16 WMMA (v_wmma_f32_16x16x32_bf16) for the LSTM matvecs,
// fp32 VALU for logits/softmax/feedback.
// ---------------------------------------------------------------------------

#define LNUM   2
#define HDIM   1024
#define G4H    4096
#define TSTEPS 64
#define NWG    32
#define TPB    256
#define WAVES_PER_WG (TPB / 32)          // 8 waves (wave32)
#define NWAVES (NWG * WAVES_PER_WG)      // 256 waves == 4096/16 output tiles

typedef __attribute__((ext_vector_type(16))) __bf16 v16bf;
typedef __attribute__((ext_vector_type(8)))  __bf16 v8bf;
typedef __attribute__((ext_vector_type(8)))  float  v8f;

struct ControllerState {
    unsigned bar_count;
    unsigned bar_phase;
    unsigned pad[62];                    // 256B barrier block
    float h[LNUM][HDIM];
    float c[LNUM][HDIM];
    float gates[G4H];
    float logits[HDIM];
    float dec[HDIM];
    __attribute__((aligned(32))) __bf16 xb[HDIM];
    __attribute__((aligned(32))) __bf16 hb[LNUM][HDIM];
};

__device__ __forceinline__ __bf16 f2bf(float f) {
    union { float f; unsigned u; } v; v.f = f;
    unsigned u = v.u;
    unsigned short s = (unsigned short)((u + 0x7FFFu + ((u >> 16) & 1u)) >> 16); // RNE
    union { unsigned short s; __bf16 b; } o; o.s = s;
    return o.b;
}

__device__ __forceinline__ float sigm(float x) { return 1.0f / (1.0f + __expf(-x)); }

// --- fp32 -> bf16 weight conversion (once per launch; feeds L2-resident set)
__global__ __launch_bounds__(TPB) void cvt_bf16_kernel(const float* __restrict__ a,
                                                       const float* __restrict__ b,
                                                       __bf16* __restrict__ oa,
                                                       __bf16* __restrict__ ob) {
    size_t i = ((size_t)blockIdx.x * TPB + threadIdx.x) * 4;
    float4 va = *(const float4*)(a + i);
    float4 vb = *(const float4*)(b + i);
    oa[i + 0] = f2bf(va.x); oa[i + 1] = f2bf(va.y);
    oa[i + 2] = f2bf(va.z); oa[i + 3] = f2bf(va.w);
    ob[i + 0] = f2bf(vb.x); ob[i + 1] = f2bf(vb.y);
    ob[i + 2] = f2bf(vb.z); ob[i + 3] = f2bf(vb.w);
}

// --- device-wide sense barrier (32 co-resident WGs)
__device__ __forceinline__ void grid_barrier(ControllerState* S, unsigned ep) {
    __syncthreads();
    if (threadIdx.x == 0) {
        __builtin_amdgcn_fence(__ATOMIC_RELEASE, "agent");
        unsigned prev = __hip_atomic_fetch_add(&S->bar_count, 1u, __ATOMIC_RELAXED,
                                               __HIP_MEMORY_SCOPE_AGENT);
        if (prev == NWG - 1u) {
            __hip_atomic_store(&S->bar_count, 0u, __ATOMIC_RELAXED, __HIP_MEMORY_SCOPE_AGENT);
            __hip_atomic_fetch_add(&S->bar_phase, 1u, __ATOMIC_RELEASE, __HIP_MEMORY_SCOPE_AGENT);
        } else {
            while (__hip_atomic_load(&S->bar_phase, __ATOMIC_ACQUIRE,
                                     __HIP_MEMORY_SCOPE_AGENT) < ep) {
                __builtin_amdgcn_s_sleep(2);
            }
        }
    }
    __syncthreads();
    __builtin_amdgcn_fence(__ATOMIC_ACQUIRE, "agent");
}

// A fragment: input vector broadcast into all 16 rows of the 16x32 bf16 A tile.
// Contiguous k order used here and for B: lanes 0-15 hold K={0..15}, lanes 16-31 K={16..31}.
__device__ __forceinline__ v16bf make_a(const __bf16* v, int kt, int lane) {
    const int base = (lane & 16) ? 16 : 0;
    const __bf16* p = v + kt * 32 + base;
    v8bf lo = *(const v8bf*)(p);
    v8bf hi = *(const v8bf*)(p + 8);
    return __builtin_shufflevector(lo, hi, 0, 1, 2, 3, 4, 5, 6, 7,
                                            8, 9, 10, 11, 12, 13, 14, 15);
}

__device__ __forceinline__ v8f wmma_bf16(v16bf a, v16bf b, v8f c) {
    return __builtin_amdgcn_wmma_f32_16x16x32_bf16(false, a, false, b,
                                                   (short)0, c, false, false);
}

__global__ __launch_bounds__(TPB) void controller_persistent(
    const float* __restrict__ bih, const float* __restrict__ bhh,
    const float* __restrict__ We,  const float* __restrict__ be,
    const __bf16* __restrict__ WihB, const __bf16* __restrict__ WhhB,
    ControllerState* __restrict__ S, float* __restrict__ out) {

    const int tid  = threadIdx.x;
    const int lane = tid & 31;
    const int wave = blockIdx.x * WAVES_PER_WG + (tid >> 5);   // 0..255
    unsigned ep = 0;

    __shared__ __align__(32) __bf16 sx[HDIM];   // layer input (bf16)
    __shared__ __align__(32) __bf16 sh[HDIM];   // recurrent h  (bf16)
    __shared__ float shf[HDIM];                 // top hidden (fp32) for logits
    __shared__ float sred[TPB];

    for (int t = 0; t < TSTEPS; ++t) {
        const float* Wet = We + (size_t)t * HDIM * HDIM;

        // ---------------- LSTM layers: gates via bf16 WMMA matvec ----------
        for (int l = 0; l < LNUM; ++l) {
            const __bf16* inv = (l == 0) ? S->xb : S->hb[0];
            const __bf16* hv  = S->hb[l];
            for (int i = tid; i < HDIM; i += TPB) { sx[i] = inv[i]; sh[i] = hv[i]; }
            __syncthreads();

            // wave owns output tile jt = wave: rows [16*wave, 16*wave+16)
            const size_t lofs = (size_t)l * G4H * HDIM;
            const size_t rsel = (size_t)(wave * 16 + (lane & 15)) * HDIM + ((lane >> 4) << 4);
            const __bf16* rowI = WihB + lofs + rsel;
            const __bf16* rowH = WhhB + lofs + rsel;

            v8f acc = {0.f, 0.f, 0.f, 0.f, 0.f, 0.f, 0.f, 0.f};
            #pragma unroll 4
            for (int kt = 0; kt < HDIM / 32; ++kt) {           // x @ W_ih^T
                v16bf a = make_a(sx, kt, lane);
                v16bf b = *(const v16bf*)(rowI + kt * 32);
                acc = wmma_bf16(a, b, acc);
            }
            #pragma unroll 4
            for (int kt = 0; kt < HDIM / 32; ++kt) {           // h @ W_hh^T
                v16bf a = make_a(sh, kt, lane);
                v16bf b = *(const v16bf*)(rowH + kt * 32);
                acc = wmma_bf16(a, b, acc);
            }
            // D row 0: lane n<16 holds gates[16*wave + n]
            if (lane < 16) S->gates[wave * 16 + lane] = acc[0];
            grid_barrier(S, ++ep);

            // ---- elementwise cell update (i,f,g,o gate order) ----
            int n = blockIdx.x * TPB + tid;
            if (n < HDIM) {
                const float* bi = bih + l * G4H;
                const float* bh = bhh + l * G4H;
                float gi = S->gates[n]            + bi[n]            + bh[n];
                float gf = S->gates[HDIM + n]     + bi[HDIM + n]     + bh[HDIM + n];
                float gg = S->gates[2 * HDIM + n] + bi[2 * HDIM + n] + bh[2 * HDIM + n];
                float go = S->gates[3 * HDIM + n] + bi[3 * HDIM + n] + bh[3 * HDIM + n];
                float cn = sigm(gf) * S->c[l][n] + sigm(gi) * tanhf(gg);
                float hn = sigm(go) * tanhf(cn);
                S->c[l][n] = cn;
                S->h[l][n] = hn;
                S->hb[l][n] = f2bf(hn);
            }
            grid_barrier(S, ++ep);
        }

        // ---------------- logits = h_top @ We_t^T + be_t (fp32) ------------
        for (int i = tid; i < HDIM; i += TPB) shf[i] = S->h[1][i];
        __syncthreads();
        for (int jj = 0; jj < 4; ++jj) {                        // 4 rows / wave
            int j = wave * 4 + jj;
            const float* row = Wet + (size_t)j * HDIM;
            __builtin_prefetch(row + HDIM, 0, 0);               // global_prefetch_b8
            float s = 0.f;
            #pragma unroll
            for (int it = 0; it < HDIM / 128; ++it) {
                int k = it * 128 + lane * 4;
                float4 wv = *(const float4*)(row + k);
                s += wv.x * shf[k] + wv.y * shf[k + 1] + wv.z * shf[k + 2] + wv.w * shf[k + 3];
            }
            for (int off = 16; off; off >>= 1) s += __shfl_xor(s, off, 32);
            if (lane == 0) S->logits[j] = s + be[(size_t)t * HDIM + j];
        }
        grid_barrier(S, ++ep);

        // ---------------- softmax (WG0) ------------------------------------
        if (blockIdx.x == 0) {
            float m = -1e30f;
            for (int i = tid; i < HDIM; i += TPB) m = fmaxf(m, S->logits[i]);
            sred[tid] = m; __syncthreads();
            for (int s2 = TPB / 2; s2; s2 >>= 1) {
                if (tid < s2) sred[tid] = fmaxf(sred[tid], sred[tid + s2]);
                __syncthreads();
            }
            float mx = sred[0]; __syncthreads();
            float ssum = 0.f;
            for (int i = tid; i < HDIM; i += TPB) {
                float e = __expf(S->logits[i] - mx);
                S->dec[i] = e; ssum += e;
            }
            sred[tid] = ssum; __syncthreads();
            for (int s2 = TPB / 2; s2; s2 >>= 1) {
                if (tid < s2) sred[tid] += sred[tid + s2];
                __syncthreads();
            }
            float invs = 1.f / sred[0]; __syncthreads();
            for (int i = tid; i < HDIM; i += TPB) {
                float d = S->dec[i] * invs;
                S->dec[i] = d;
                out[(size_t)t * HDIM + i] = d;
            }
        }
        grid_barrier(S, ++ep);

        // ---------------- x_next = dec @ We_t (deterministic, WGs 0..3) ----
        if (blockIdx.x < 4) {
            int hIdx = blockIdx.x * TPB + tid;
            float acc2 = 0.f;
            for (int j = 0; j < HDIM; j += 4) {
                float d0 = S->dec[j], d1 = S->dec[j + 1];
                float d2 = S->dec[j + 2], d3 = S->dec[j + 3];
                acc2 += d0 * Wet[(size_t)j * HDIM + hIdx]
                      + d1 * Wet[(size_t)(j + 1) * HDIM + hIdx]
                      + d2 * Wet[(size_t)(j + 2) * HDIM + hIdx]
                      + d3 * Wet[(size_t)(j + 3) * HDIM + hIdx];
            }
            S->xb[hIdx] = f2bf(acc2);
        }
        grid_barrier(S, ++ep);
    }
}

extern "C" void kernel_launch(void* const* d_in, const int* in_sizes, int n_in,
                              void* d_out, int out_size, void* d_ws, size_t ws_size,
                              hipStream_t stream) {
    (void)in_sizes; (void)n_in; (void)out_size; (void)ws_size;
    const float* W_ih = (const float*)d_in[0];   // [L, 4H, H]
    const float* W_hh = (const float*)d_in[1];   // [L, 4H, H]
    const float* b_ih = (const float*)d_in[2];   // [L, 4H]
    const float* b_hh = (const float*)d_in[3];   // [L, 4H]
    const float* We   = (const float*)d_in[4];   // [T, E, H]
    const float* be   = (const float*)d_in[5];   // [T, E]
    float* out = (float*)d_out;                  // [T, E]

    const size_t nW = (size_t)LNUM * G4H * HDIM; // 8,388,608 elements
    char* p = (char*)d_ws;
    __bf16* WihB = (__bf16*)p; p += nW * sizeof(unsigned short);
    __bf16* WhhB = (__bf16*)p; p += nW * sizeof(unsigned short);
    p = (char*)(((uintptr_t)p + 255) & ~(uintptr_t)255);
    ControllerState* S = (ControllerState*)p;

    // zero barrier counters + h/c/x state (graph-capture-safe)
    hipMemsetAsync(S, 0, sizeof(ControllerState), stream);

    const int cvtBlocks = (int)(nW / 4 / TPB);   // 4 elems/thread, exact divide
    cvt_bf16_kernel<<<cvtBlocks, TPB, 0, stream>>>(W_ih, W_hh, WihB, WhhB);

    controller_persistent<<<NWG, TPB, 0, stream>>>(b_ih, b_hh, We, be,
                                                   WihB, WhhB, S, out);
}
